// VAE_12309376270724
// MI455X (gfx1250) — compile-verified, matrix operand order checked by reading, not run on hardware
//
#include <hip/hip_runtime.h>

// ---------------------------------------------------------------------------
// VAE forward for MI455X (gfx1250, wave32, WMMA).
// All six GEMMs run as bf16 WMMA (v_wmma_f32_16x16x32_bf16) with f32
// accumulation. Tiles are staged with CDNA5 async global->LDS copies
// (global_load_async_to_lds_b128, ASYNCcnt) into a double-buffered LDS
// pipeline, so next-tile loads overlap the WMMA burst.
// Dims padded to multiples of 128 via zero-padded transposed bf16 weight
// copies so the GEMM inner loop is branch-free.
// Workspace requirement: ~188 MB.
// d_out layout (fp32): [loss, loss_recon, loss_kl, x_recon(4096x4096), z(4096x256)]
// ---------------------------------------------------------------------------

typedef __attribute__((ext_vector_type(16))) __bf16 v16bf;
typedef __attribute__((ext_vector_type(8)))  float  v8f;

union FragAB { v16bf v; uint4 q[2]; };

__device__ __forceinline__ unsigned short f2bf(float f) {
  unsigned int u = __float_as_uint(f);
  unsigned int r = u + 0x7FFFu + ((u >> 16) & 1u);   // round-to-nearest-even
  return (unsigned short)(r >> 16);
}

// ---- CDNA5 async global->LDS (ASYNCcnt) -----------------------------------
// One instruction moves 16B per lane directly into LDS, no VGPR staging.
// LDS operand = low 32 bits of the generic shared pointer (how AS(3) lowers).
__device__ __forceinline__ void async_ld16(const unsigned short* g, unsigned short* l) {
  unsigned int lds = (unsigned int)(uintptr_t)l;
  unsigned long long ga = (unsigned long long)(uintptr_t)g;
  asm volatile("global_load_async_to_lds_b128 %0, %1, off"
               :: "v"(lds), "v"(ga)
               : "memory");
}

template <int N>
__device__ __forceinline__ void wait_async() {
#if __has_builtin(__builtin_amdgcn_s_wait_asynccnt)
  __builtin_amdgcn_s_wait_asynccnt((unsigned short)N);
#else
  asm volatile("s_wait_asynccnt %0" :: "n"(N) : "memory");
#endif
}

// --------------------------- conversion kernels ----------------------------

__global__ __launch_bounds__(256)
void cvt_x_kernel(const float* __restrict__ x, unsigned short* __restrict__ o, int n) {
  int i = blockIdx.x * 256 + threadIdx.x;
  if (i < n) o[i] = f2bf(x[i]);
}

// W is [Ko, No] row-major fp32; produce Wt [Np, Kp] row-major bf16 (transposed,
// zero padded). Padded rows/cols are zero so padded outputs become silu(0)=0.
__global__ __launch_bounds__(256)
void cvt_wt_kernel(const float* __restrict__ W, unsigned short* __restrict__ Wt,
                   int Ko, int No, int Kp, int Np) {
  int i = blockIdx.x * 256 + threadIdx.x;
  int tot = Np * Kp;
  if (i >= tot) return;
  int n = i / Kp;
  int k = i - n * Kp;
  float v = (n < No && k < Ko) ? W[(size_t)k * No + n] : 0.0f;
  Wt[i] = f2bf(v);
}

// ------------------------------- WMMA GEMM ---------------------------------
// C[M,Np] = A[M,K](bf16) * Wt[Np,K]^T (bf16) + bias, fused epilogue.
// MODE 0: out = bf16(silu(v))      MODE 1: out = fp32 v
// MODE 2: x_recon = sigmoid(v); per-block BCE partial sums (deterministic).
// Block: 256 threads (8 waves), tile 128x128, K-step 32, double-buffered LDS
// filled by async global->LDS copies (4 async instrs per wave per tile).
// Wave grid 4x2 -> each wave: 32(M) x 64(N) = 2x4 fragments of 16x16.

template <int MODE>
__global__ __launch_bounds__(256)
void gemm_wmma_kernel(const unsigned short* __restrict__ A,
                      const unsigned short* __restrict__ Bt,
                      const float* __restrict__ bias,
                      int K, int Np, int nbias,
                      unsigned short* __restrict__ outb,
                      float* __restrict__ outf,
                      const float* __restrict__ Xref,
                      float* __restrict__ xrecon,
                      float* __restrict__ part) {
  __shared__ unsigned short lA[2][128][32];   // 2 x 8 KB, M-major
  __shared__ unsigned short lB[2][128][32];   // 2 x 8 KB, N-major (Wt rows)
  __shared__ float red[256];

  const int m0   = blockIdx.y * 128;
  const int n0   = blockIdx.x * 128;
  const int w    = threadIdx.x >> 5;
  const int lane = threadIdx.x & 31;
  const int wm   = (w >> 1) * 32;   // wave M offset in tile
  const int wn   = (w & 1) * 64;    // wave N offset in tile
  const int l15  = lane & 15;
  const int kh   = lane >> 4;       // K-half select per ISA 16-bit A layout

  // Per-thread staging coordinates: 256 threads x 2 x 16B = 8 KB per operand.
  const int c0   = threadIdx.x * 2;
  const int row0 = c0 >> 2, col0 = (c0 & 3) * 8;
  const int row1 = (c0 + 1) >> 2, col1 = ((c0 + 1) & 3) * 8;

  v8f acc[2][4] = {};

  // Issue async loads for a k-tile into LDS buffer `buf` (4 instrs / wave).
  auto issue_tile = [&](int buf, int k0) {
    async_ld16(A  + (size_t)(m0 + row0) * K + (k0 + col0), &lA[buf][row0][col0]);
    async_ld16(Bt + (size_t)(n0 + row0) * K + (k0 + col0), &lB[buf][row0][col0]);
    async_ld16(A  + (size_t)(m0 + row1) * K + (k0 + col1), &lA[buf][row1][col1]);
    async_ld16(Bt + (size_t)(n0 + row1) * K + (k0 + col1), &lB[buf][row1][col1]);
  };

  issue_tile(0, 0);
  int buf = 0;

  for (int k0 = 0; k0 < K; k0 += 32) {
    const bool have_next = (k0 + 32) < K;
    // All waves finished reading buf^1 (computed on last iteration) before we
    // overwrite it with the next tile's async copies.
    __syncthreads();
    if (have_next) {
      issue_tile(buf ^ 1, k0 + 32);
      wait_async<4>();   // retire this wave's 4 in-order copies for tile `buf`
    } else {
      wait_async<0>();
    }
    __syncthreads();     // all waves' slices of tile `buf` landed in LDS

    // Fragment loads per ISA layout: lanes 0-15 take K{0..7,16..23},
    // lanes 16-31 take K{8..15,24..31}; 2x b128 LDS reads per fragment.
    FragAB a[2], b[4];
    #pragma unroll
    for (int fm = 0; fm < 2; ++fm) {
      int r = wm + fm * 16 + l15;
      a[fm].q[0] = *(const uint4*)(&lA[buf][r][kh * 8]);
      a[fm].q[1] = *(const uint4*)(&lA[buf][r][16 + kh * 8]);
    }
    #pragma unroll
    for (int fn = 0; fn < 4; ++fn) {
      int cidx = wn + fn * 16 + l15;
      b[fn].q[0] = *(const uint4*)(&lB[buf][cidx][kh * 8]);
      b[fn].q[1] = *(const uint4*)(&lB[buf][cidx][16 + kh * 8]);
    }
    #pragma unroll
    for (int fm = 0; fm < 2; ++fm)
      #pragma unroll
      for (int fn = 0; fn < 4; ++fn)
        acc[fm][fn] = __builtin_amdgcn_wmma_f32_16x16x32_bf16(
            false, a[fm].v, false, b[fn].v, (short)0, acc[fm][fn], false, false);

    buf ^= 1;
  }

  // Epilogue. C/D layout: VGPR r, lanes 0-15 -> M=r, lanes 16-31 -> M=8+r.
  float lsum = 0.0f;
  #pragma unroll
  for (int fm = 0; fm < 2; ++fm) {
    #pragma unroll
    for (int fn = 0; fn < 4; ++fn) {
      int col  = n0 + wn + fn * 16 + l15;
      int rowb = m0 + wm + fm * 16 + kh * 8;
      float bn = (col < nbias) ? bias[col] : 0.0f;
      #pragma unroll
      for (int r = 0; r < 8; ++r) {
        float v  = acc[fm][fn][r] + bn;
        int row  = rowb + r;
        if (MODE == 0) {
          float s = v / (1.0f + __expf(-v));            // SiLU
          outb[(size_t)row * Np + col] = f2bf(s);
        } else if (MODE == 1) {
          outf[(size_t)row * Np + col] = v;
        } else {
          float xr = 1.0f / (1.0f + __expf(-v));        // sigmoid
          xrecon[(size_t)row * Np + col] = xr;
          float xv = Xref[(size_t)row * Np + col];
          lsum += fmaxf(v, 0.0f) - v * xv + log1pf(__expf(-fabsf(v)));
        }
      }
    }
  }
  if (MODE == 2) {
    red[threadIdx.x] = lsum;
    __syncthreads();
    for (int s = 128; s > 0; s >>= 1) {
      if (threadIdx.x < s) red[threadIdx.x] += red[threadIdx.x + s];
      __syncthreads();
    }
    if (threadIdx.x == 0) part[blockIdx.y * gridDim.x + blockIdx.x] = red[0];
  }
}

// ------------------------------ reparam + KL -------------------------------

__global__ __launch_bounds__(256)
void reparam_kernel(const float* __restrict__ enc, const float* __restrict__ noise,
                    float* __restrict__ zout, unsigned short* __restrict__ zb,
                    float* __restrict__ kl_part) {
  __shared__ float red[256];
  int i   = blockIdx.x * 256 + threadIdx.x;   // over 4096*256
  int row = i >> 8;
  int col = i & 255;
  float mu = enc[(size_t)row * 512 + col];
  float lv = enc[(size_t)row * 512 + 256 + col];
  float sp = (lv > 15.0f) ? lv : log1pf(__expf(lv));  // softplus
  float sc = sp + 1e-8f;
  float z  = fmaf(sc, noise[i], mu);
  zout[i] = z;
  zb[i]   = f2bf(z);
  float kl = 0.5f * (sc * sc + mu * mu - 1.0f - 2.0f * logf(sc));
  red[threadIdx.x] = kl;
  __syncthreads();
  for (int s = 128; s > 0; s >>= 1) {
    if (threadIdx.x < s) red[threadIdx.x] += red[threadIdx.x + s];
    __syncthreads();
  }
  if (threadIdx.x == 0) kl_part[blockIdx.x] = red[0];
}

// ------------------------- deterministic finalize --------------------------

__global__ __launch_bounds__(256)
void finalize_kernel(const float* __restrict__ bce_part, int nb,
                     const float* __restrict__ kl_part, int nk,
                     float* __restrict__ out) {
  __shared__ float red[256];
  float a = 0.0f;
  for (int i = threadIdx.x; i < nb; i += 256) a += bce_part[i];
  red[threadIdx.x] = a;
  __syncthreads();
  for (int s = 128; s > 0; s >>= 1) {
    if (threadIdx.x < s) red[threadIdx.x] += red[threadIdx.x + s];
    __syncthreads();
  }
  float bce_sum = red[0];
  __syncthreads();
  float b = 0.0f;
  for (int i = threadIdx.x; i < nk; i += 256) b += kl_part[i];
  red[threadIdx.x] = b;
  __syncthreads();
  for (int s = 128; s > 0; s >>= 1) {
    if (threadIdx.x < s) red[threadIdx.x] += red[threadIdx.x + s];
    __syncthreads();
  }
  if (threadIdx.x == 0) {
    float kl_sum = red[0];
    float lr = bce_sum / 4096.0f;
    float lk = 1.0f * (kl_sum / 4096.0f);   // BETA = 1
    out[0] = lr + lk;
    out[1] = lr;
    out[2] = lk;
  }
}

// -------------------------------- launcher ---------------------------------

extern "C" void kernel_launch(void* const* d_in, const int* in_sizes, int n_in,
                              void* d_out, int out_size, void* d_ws, size_t ws_size,
                              hipStream_t stream) {
  const float* x     = (const float*)d_in[0];
  const float* noise = (const float*)d_in[1];
  const float* ew0   = (const float*)d_in[2];
  const float* eb0   = (const float*)d_in[3];
  const float* ew1   = (const float*)d_in[4];
  const float* eb1   = (const float*)d_in[5];
  const float* ew2   = (const float*)d_in[6];
  const float* eb2   = (const float*)d_in[7];
  const float* dw0   = (const float*)d_in[8];
  const float* db0   = (const float*)d_in[9];
  const float* dw1   = (const float*)d_in[10];
  const float* db1   = (const float*)d_in[11];
  const float* dw2   = (const float*)d_in[12];
  const float* db2   = (const float*)d_in[13];

  char* ws = (char*)d_ws;
  size_t cur = 0;
  auto carve = [&](size_t bytes) -> char* {
    char* p = ws + cur;
    cur += (bytes + 255) & ~(size_t)255;
    return p;
  };

  // bf16 activations / weights (padded dims: 4096,2944,1792,512 | 256,1536,2816,4096)
  unsigned short* xb   = (unsigned short*)carve((size_t)4096 * 4096 * 2);
  unsigned short* ew0t = (unsigned short*)carve((size_t)2944 * 4096 * 2);
  unsigned short* ew1t = (unsigned short*)carve((size_t)1792 * 2944 * 2);
  unsigned short* ew2t = (unsigned short*)carve((size_t)512  * 1792 * 2);
  unsigned short* dw0t = (unsigned short*)carve((size_t)1536 * 256  * 2);
  unsigned short* dw1t = (unsigned short*)carve((size_t)2816 * 1536 * 2);
  unsigned short* dw2t = (unsigned short*)carve((size_t)4096 * 2816 * 2);
  unsigned short* h0   = (unsigned short*)carve((size_t)4096 * 2944 * 2);
  unsigned short* h1   = (unsigned short*)carve((size_t)4096 * 1792 * 2);
  float*          enc  = (float*)carve((size_t)4096 * 512 * 4);
  unsigned short* zb   = (unsigned short*)carve((size_t)4096 * 256 * 2);
  unsigned short* hd0  = (unsigned short*)carve((size_t)4096 * 1536 * 2);
  unsigned short* hd1  = (unsigned short*)carve((size_t)4096 * 2816 * 2);
  float*          bce_part = (float*)carve(1024 * 4);
  float*          kl_part  = (float*)carve(4096 * 4);

  float* out    = (float*)d_out;
  float* xrecon = out + 3;
  float* zout   = out + 3 + (size_t)4096 * 4096;

  auto cdiv = [](int a, int b) { return (a + b - 1) / b; };
  dim3 blk(256);

  // fp32 -> bf16 conversions (one-shot, bandwidth-trivial vs 23.3 TB/s)
  cvt_x_kernel<<<cdiv(4096 * 4096, 256), blk, 0, stream>>>(x, xb, 4096 * 4096);
  cvt_wt_kernel<<<cdiv(2944 * 4096, 256), blk, 0, stream>>>(ew0, ew0t, 4096, 2901, 4096, 2944);
  cvt_wt_kernel<<<cdiv(1792 * 2944, 256), blk, 0, stream>>>(ew1, ew1t, 2901, 1707, 2944, 1792);
  cvt_wt_kernel<<<cdiv(512  * 1792, 256), blk, 0, stream>>>(ew2, ew2t, 1707, 512,  1792, 512);
  cvt_wt_kernel<<<cdiv(1536 * 256,  256), blk, 0, stream>>>(dw0, dw0t, 256,  1536, 256,  1536);
  cvt_wt_kernel<<<cdiv(2816 * 1536, 256), blk, 0, stream>>>(dw1, dw1t, 1536, 2816, 1536, 2816);
  cvt_wt_kernel<<<cdiv(4096 * 2816, 256), blk, 0, stream>>>(dw2, dw2t, 2816, 4096, 2816, 4096);

  // Encoder
  gemm_wmma_kernel<0><<<dim3(2944 / 128, 32), blk, 0, stream>>>(
      xb, ew0t, eb0, 4096, 2944, 2901, h0, nullptr, nullptr, nullptr, nullptr);
  gemm_wmma_kernel<0><<<dim3(1792 / 128, 32), blk, 0, stream>>>(
      h0, ew1t, eb1, 2944, 1792, 1707, h1, nullptr, nullptr, nullptr, nullptr);
  gemm_wmma_kernel<1><<<dim3(512 / 128, 32), blk, 0, stream>>>(
      h1, ew2t, eb2, 1792, 512, 512, nullptr, enc, nullptr, nullptr, nullptr);

  // Reparameterize + KL partials + z outputs (fp32 to d_out, bf16 for decoder)
  reparam_kernel<<<4096, blk, 0, stream>>>(enc, noise, zout, zb, kl_part);

  // Decoder
  gemm_wmma_kernel<0><<<dim3(1536 / 128, 32), blk, 0, stream>>>(
      zb, dw0t, db0, 256, 1536, 1536, hd0, nullptr, nullptr, nullptr, nullptr);
  gemm_wmma_kernel<0><<<dim3(2816 / 128, 32), blk, 0, stream>>>(
      hd0, dw1t, db1, 1536, 2816, 2816, hd1, nullptr, nullptr, nullptr, nullptr);
  gemm_wmma_kernel<2><<<dim3(4096 / 128, 32), blk, 0, stream>>>(
      hd1, dw2t, db2, 2816, 4096, 4096, nullptr, nullptr, x, xrecon, bce_part);

  // Deterministic scalar losses
  finalize_kernel<<<1, blk, 0, stream>>>(bce_part, 1024, kl_part, 4096, out);
}
